// SlidingWindowGQA_56642028699979
// MI455X (gfx1250) — matrix operand order repaired and checked by reading, not compiled, not run on hardware
//
#include <hip/hip_runtime.h>

// Problem constants (fixed by the reference).
#define B_SZ   2
#define T_SZ   2048
#define D_SZ   1024
#define H_SZ   16
#define KV_SZ  4
#define DH_SZ  64
#define WIN_SZ 512
#define SCALE_F 0.125f   // 1/sqrt(64)

typedef __attribute__((ext_vector_type(16))) __bf16        v16bf;
typedef __attribute__((ext_vector_type(8)))  float         v8f;
typedef __attribute__((ext_vector_type(8)))  unsigned int  v8u;

static __device__ __forceinline__ unsigned short f2bf(float f) {
  unsigned int u = __float_as_uint(f);
  return (unsigned short)((u + 0x7FFFu + ((u >> 16) & 1u)) >> 16);  // RNE
}
static __device__ __forceinline__ unsigned int pkbf(float lo, float hi) {
  return (unsigned int)f2bf(lo) | ((unsigned int)f2bf(hi) << 16);
}
static __device__ __forceinline__ v8f wmma_bf16f32(v8u a, v8u b, v8f c) {
  return __builtin_amdgcn_wmma_f32_16x16x32_bf16(
      false, __builtin_bit_cast(v16bf, a),
      false, __builtin_bit_cast(v16bf, b),
      (short)0, c, false, false);
}

// ---------------------------------------------------------------------------
// Kernel 1: Y = X(f32) @ W^T + bias, optional RoPE, bf16 output.
// One wave computes a 16x16 output tile; K loop in steps of 32 (1 WMMA/step).
// A fragment: lane l holds row (l&15), K = kb + (l>>4)*8 + {0..7, 16..23}.
// B fragment: lane l holds col (l&15) (= row of W), K = kb + (l>>4)*8 + {0..7, 16..23}.
// storeVT=1 stores V transposed: Ob[((row/T)*256 + n)*T + row%T].
// ---------------------------------------------------------------------------
__global__ void __launch_bounds__(256)
qkv_proj_kernel(const float* __restrict__ X, const float* __restrict__ W,
                const float* __restrict__ bias,
                const float* __restrict__ rc, const float* __restrict__ rs,
                unsigned short* __restrict__ Ob,
                int M, int N, int Kd, int applyRope, int storeVT)
{
  const int wave   = blockIdx.x * (blockDim.x >> 5) + (threadIdx.x >> 5);
  const int ntiles = N >> 4;
  const int mt = wave / ntiles, nt = wave % ntiles;
  if (mt >= (M >> 4)) return;
  const int lane = threadIdx.x & 31;
  const int half = lane >> 4, col = lane & 15;
  const int mbase = mt << 4, nbase = nt << 4;

  const float* arow = X + (size_t)(mbase + col) * Kd + half * 8;
  const float* brow = W + (size_t)(nbase + col) * Kd + half * 8;

  v8f c = {};
  for (int kb = 0; kb < Kd; kb += 32) {
    const float4* ap = reinterpret_cast<const float4*>(arow + kb);
    float4 a0 = ap[0], a1 = ap[1];   // K = base+0..7
    float4 a2 = ap[4], a3 = ap[5];   // K = base+16..23
    v8u au;
    au[0]=pkbf(a0.x,a0.y); au[1]=pkbf(a0.z,a0.w);
    au[2]=pkbf(a1.x,a1.y); au[3]=pkbf(a1.z,a1.w);
    au[4]=pkbf(a2.x,a2.y); au[5]=pkbf(a2.z,a2.w);
    au[6]=pkbf(a3.x,a3.y); au[7]=pkbf(a3.z,a3.w);

    const float4* bp = reinterpret_cast<const float4*>(brow + kb);
    float4 b0 = bp[0], b1 = bp[1], b2 = bp[4], b3 = bp[5];
    v8u bu;
    bu[0]=pkbf(b0.x,b0.y); bu[1]=pkbf(b0.z,b0.w);
    bu[2]=pkbf(b1.x,b1.y); bu[3]=pkbf(b1.z,b1.w);
    bu[4]=pkbf(b2.x,b2.y); bu[5]=pkbf(b2.z,b2.w);
    bu[6]=pkbf(b3.x,b3.y); bu[7]=pkbf(b3.z,b3.w);

    c = wmma_bf16f32(au, bu, c);
  }

  const int n = nbase + col;
  const float bv = bias[n];
  float vals[8];
#pragma unroll
  for (int r = 0; r < 8; ++r) vals[r] = c[r] + bv;

  if (applyRope) {
    const int p = (n & (DH_SZ - 1)) >> 1;   // rotary pair index within head
#pragma unroll
    for (int r = 0; r < 8; ++r) {
      const int row = mbase + r + half * 8;
      const int t   = row & (T_SZ - 1);
      const float cv = rc[t * (DH_SZ / 2) + p];
      const float sv = rs[t * (DH_SZ / 2) + p];
      const float partner = __shfl_xor(vals[r], 1, 32);  // even<->odd feature
      vals[r] = (n & 1) ? __fmaf_rn(partner, sv,  vals[r] * cv)   // xo' = xe*s + xo*c
                        : __fmaf_rn(vals[r], cv, -partner * sv);  // xe' = xe*c - xo*s
    }
  }

#pragma unroll
  for (int r = 0; r < 8; ++r) {
    const int row = mbase + r + half * 8;
    const unsigned short hb = f2bf(vals[r]);
    if (storeVT)
      Ob[(size_t)((row >> 11) * (KV_SZ * DH_SZ) + n) * T_SZ + (row & (T_SZ - 1))] = hb;
    else
      Ob[(size_t)row * N + n] = hb;
  }
}

// ---------------------------------------------------------------------------
// Kernel 2: sliding-window GQA flash attention; one wave per (b, h, 16 queries).
// ---------------------------------------------------------------------------
__global__ void __launch_bounds__(256)
attn_kernel(const unsigned short* __restrict__ Qb,
            const unsigned short* __restrict__ Kb,
            const unsigned short* __restrict__ Vt,
            unsigned short* __restrict__ Ob)
{
  __shared__ unsigned short psm[8 * 512];   // per-wave 16x32 bf16 P tile
  const int wv   = threadIdx.x >> 5;
  const int wave = blockIdx.x * 8 + wv;
  const int qt = wave & (T_SZ / 16 - 1);
  const int h  = (wave >> 7) & (H_SZ - 1);
  const int b  = wave >> 11;
  const int kv = h >> 2;                    // G = 4
  const int lane = threadIdx.x & 31;
  const int half = lane >> 4, col = lane & 15;
  const int qbase = qt << 4;
  unsigned short* pb = psm + wv * 512;

  // Q A-fragments: rows = 16 queries, K-dim = DH=64 split into two K=32 frags.
  const uint4* q4 = reinterpret_cast<const uint4*>(
      Qb + ((size_t)(b * T_SZ + qbase + col) << 10) + h * DH_SZ + half * 16);
  uint4 t0 = q4[0], t1 = q4[1], t2 = q4[4], t3 = q4[5];
  v8u qa0, qa1;
  qa0[0]=t0.x; qa0[1]=t0.y; qa0[2]=t0.z; qa0[3]=t0.w;
  qa0[4]=t1.x; qa0[5]=t1.y; qa0[6]=t1.z; qa0[7]=t1.w;
  qa1[0]=t2.x; qa1[1]=t2.y; qa1[2]=t2.z; qa1[3]=t2.w;
  qa1[4]=t3.x; qa1[5]=t3.y; qa1[6]=t3.z; qa1[7]=t3.w;

  v8f o[4] = {};
  float mrow[8], lrow[8];
#pragma unroll
  for (int r = 0; r < 8; ++r) { mrow[r] = -3.0e38f; lrow[r] = 0.0f; }

  int kb_start = qbase - (WIN_SZ - 1);
  if (kb_start < 0) kb_start = 0;
  kb_start &= ~31;
  const int qend = qbase + 16;

  for (int kb = kb_start; kb < qend; kb += 32) {
    // ---- S = Q K^T for 32 keys (two 16-key tiles) ----
    v8f s0 = {}, s1 = {};
    {
      const uint4* k4 = reinterpret_cast<const uint4*>(
          Kb + ((size_t)(b * T_SZ + kb + col) << 8) + kv * DH_SZ + half * 16);
      uint4 u0 = k4[0], u1 = k4[1], u2 = k4[2], u3 = k4[3];
      v8u bu;
      bu[0]=u0.x; bu[1]=u0.y; bu[2]=u0.z; bu[3]=u0.w;
      bu[4]=u1.x; bu[5]=u1.y; bu[6]=u1.z; bu[7]=u1.w;
      s0 = wmma_bf16f32(qa0, bu, s0);
      bu[0]=u2.x; bu[1]=u2.y; bu[2]=u2.z; bu[3]=u2.w;
      bu[4]=u3.x; bu[5]=u3.y; bu[6]=u3.z; bu[7]=u3.w;
      s0 = wmma_bf16f32(qa1, bu, s0);
    }
    {
      const uint4* k4 = reinterpret_cast<const uint4*>(
          Kb + ((size_t)(b * T_SZ + kb + 16 + col) << 8) + kv * DH_SZ + half * 16);
      uint4 u0 = k4[0], u1 = k4[1], u2 = k4[2], u3 = k4[3];
      v8u bu;
      bu[0]=u0.x; bu[1]=u0.y; bu[2]=u0.z; bu[3]=u0.w;
      bu[4]=u1.x; bu[5]=u1.y; bu[6]=u1.z; bu[7]=u1.w;
      s1 = wmma_bf16f32(qa0, bu, s1);
      bu[0]=u2.x; bu[1]=u2.y; bu[2]=u2.z; bu[3]=u2.w;
      bu[4]=u3.x; bu[5]=u3.y; bu[6]=u3.z; bu[7]=u3.w;
      s1 = wmma_bf16f32(qa1, bu, s1);
    }
    if (kb + 32 < qend)
      __builtin_prefetch(Kb + ((size_t)(b * T_SZ + kb + 32 + col) << 8) + kv * DH_SZ, 0, 3);

    // ---- mask + online softmax (row = 16-lane half; shfl_xor reductions) ----
    float p0[8], p1[8];
#pragma unroll
    for (int r = 0; r < 8; ++r) {
      const int q  = qbase + r + half * 8;
      const int k0 = kb + col;
      const int k1 = kb + 16 + col;
      const bool v0 = (k0 <= q) && (k0 > q - WIN_SZ);
      const bool v1 = (k1 <= q) && (k1 > q - WIN_SZ);
      const float x0 = v0 ? s0[r] * SCALE_F : -3.0e38f;
      const float x1 = v1 ? s1[r] * SCALE_F : -3.0e38f;
      float rm = fmaxf(x0, x1);
      rm = fmaxf(rm, __shfl_xor(rm, 1, 32));
      rm = fmaxf(rm, __shfl_xor(rm, 2, 32));
      rm = fmaxf(rm, __shfl_xor(rm, 4, 32));
      rm = fmaxf(rm, __shfl_xor(rm, 8, 32));
      const float mn = fmaxf(mrow[r], rm);
      const float e0 = v0 ? __expf(x0 - mn) : 0.0f;
      const float e1 = v1 ? __expf(x1 - mn) : 0.0f;
      float rsum = e0 + e1;
      rsum += __shfl_xor(rsum, 1, 32);
      rsum += __shfl_xor(rsum, 2, 32);
      rsum += __shfl_xor(rsum, 4, 32);
      rsum += __shfl_xor(rsum, 8, 32);
      const float alpha = __expf(mrow[r] - mn);
      lrow[r] = lrow[r] * alpha + rsum;
      mrow[r] = mn;
      o[0][r] *= alpha; o[1][r] *= alpha; o[2][r] *= alpha; o[3][r] *= alpha;
      p0[r] = e0; p1[r] = e1;
    }

    // ---- P: C-layout -> LDS -> A-layout (bf16) ----
#pragma unroll
    for (int r = 0; r < 8; ++r) {
      const int row = r + half * 8;
      pb[row * 32 + col]      = f2bf(p0[r]);
      pb[row * 32 + 16 + col] = f2bf(p1[r]);
    }
    asm volatile("s_wait_dscnt 0" ::: "memory");
    const uint4* pl = reinterpret_cast<const uint4*>(pb + col * 32 + half * 16);
    uint4 pa = pl[0], pc = pl[1];
    v8u pau;
    pau[0]=pa.x; pau[1]=pa.y; pau[2]=pa.z; pau[3]=pa.w;
    pau[4]=pc.x; pau[5]=pc.y; pau[6]=pc.z; pau[7]=pc.w;
    asm volatile("s_wait_dscnt 0" ::: "memory");

    // ---- O += P @ V (V stored transposed: rows contiguous over keys) ----
    const size_t vcol0 = ((size_t)(b * (KV_SZ * DH_SZ) + kv * DH_SZ) << 11);
#pragma unroll
    for (int j = 0; j < 4; ++j) {
      const int d = j * 16 + col;
      const uint4* v4p = reinterpret_cast<const uint4*>(
          Vt + vcol0 + ((size_t)d << 11) + kb + half * 16);
      uint4 w0 = v4p[0], w1 = v4p[1];
      v8u vu;
      vu[0]=w0.x; vu[1]=w0.y; vu[2]=w0.z; vu[3]=w0.w;
      vu[4]=w1.x; vu[5]=w1.y; vu[6]=w1.z; vu[7]=w1.w;
      o[j] = wmma_bf16f32(pau, vu, o[j]);
    }
  }

#pragma unroll
  for (int r = 0; r < 8; ++r) {
    const float inv = 1.0f / lrow[r];
    const int row = qbase + r + half * 8;
    const size_t ob = ((size_t)(b * T_SZ + row) << 10) + h * DH_SZ;
#pragma unroll
    for (int j = 0; j < 4; ++j)
      Ob[ob + j * 16 + col] = f2bf(o[j][r] * inv);
  }
}

// ---------------------------------------------------------------------------
// Kernel 3: Out(f32) = A(bf16) @ Wo^T + bo
// ---------------------------------------------------------------------------
__global__ void __launch_bounds__(256)
out_proj_kernel(const unsigned short* __restrict__ Ab,
                const float* __restrict__ W, const float* __restrict__ bias,
                float* __restrict__ Out, int M, int N, int Kd)
{
  const int wave   = blockIdx.x * (blockDim.x >> 5) + (threadIdx.x >> 5);
  const int ntiles = N >> 4;
  const int mt = wave / ntiles, nt = wave % ntiles;
  if (mt >= (M >> 4)) return;
  const int lane = threadIdx.x & 31;
  const int half = lane >> 4, col = lane & 15;
  const int mbase = mt << 4, nbase = nt << 4;

  const unsigned short* arow = Ab + (size_t)(mbase + col) * Kd + half * 16;
  const float*          brow = W  + (size_t)(nbase + col) * Kd + half * 16;

  v8f c = {};
  for (int kb = 0; kb < Kd; kb += 32) {
    const uint4* ap = reinterpret_cast<const uint4*>(arow + kb);
    uint4 al = ap[0], ah = ap[1];
    v8u au;
    au[0]=al.x; au[1]=al.y; au[2]=al.z; au[3]=al.w;
    au[4]=ah.x; au[5]=ah.y; au[6]=ah.z; au[7]=ah.w;

    const float4* bp = reinterpret_cast<const float4*>(brow + kb);
    float4 b0 = bp[0], b1 = bp[1], b2 = bp[2], b3 = bp[3];
    v8u bu;
    bu[0]=pkbf(b0.x,b0.y); bu[1]=pkbf(b0.z,b0.w);
    bu[2]=pkbf(b1.x,b1.y); bu[3]=pkbf(b1.z,b1.w);
    bu[4]=pkbf(b2.x,b2.y); bu[5]=pkbf(b2.z,b2.w);
    bu[6]=pkbf(b3.x,b3.y); bu[7]=pkbf(b3.z,b3.w);

    c = wmma_bf16f32(au, bu, c);
  }

  const float bv = bias[nbase + col];
#pragma unroll
  for (int r = 0; r < 8; ++r)
    Out[(size_t)(mbase + r + half * 8) * N + nbase + col] = c[r] + bv;
}

// ---------------------------------------------------------------------------
extern "C" void kernel_launch(void* const* d_in, const int* in_sizes, int n_in,
                              void* d_out, int out_size, void* d_ws, size_t ws_size,
                              hipStream_t stream) {
  const float* x  = (const float*)d_in[0];
  const float* Wq = (const float*)d_in[1];
  const float* bq = (const float*)d_in[2];
  const float* Wk = (const float*)d_in[3];
  const float* bk = (const float*)d_in[4];
  const float* Wv = (const float*)d_in[5];
  const float* bv = (const float*)d_in[6];
  const float* Wo = (const float*)d_in[7];
  const float* bo = (const float*)d_in[8];
  const float* rc = (const float*)d_in[9];
  const float* rs = (const float*)d_in[10];
  float* out = (float*)d_out;

  const size_t M = (size_t)B_SZ * T_SZ;          // 4096
  unsigned short* Qb = (unsigned short*)d_ws;                 // 4096x1024 bf16
  unsigned short* Kb = Qb + M * D_SZ;                         // 4096x256  bf16
  unsigned short* Vt = Kb + M * (KV_SZ * DH_SZ);              // 256x? transposed [B*256, T]
  unsigned short* Ao = Vt + M * (KV_SZ * DH_SZ);              // 4096x1024 bf16

  // Q: 256 m-tiles x 64 n-tiles = 16384 waves -> 2048 blocks of 8 waves
  qkv_proj_kernel<<<2048, 256, 0, stream>>>(x, Wq, bq, rc, rs, Qb,
                                            (int)M, D_SZ, D_SZ, 1, 0);
  // K: 256 x 16 tiles = 4096 waves -> 512 blocks
  qkv_proj_kernel<<<512, 256, 0, stream>>>(x, Wk, bk, rc, rs, Kb,
                                           (int)M, KV_SZ * DH_SZ, D_SZ, 1, 0);
  // V: no rope, stored transposed
  qkv_proj_kernel<<<512, 256, 0, stream>>>(x, Wv, bv, rc, rs, Vt,
                                           (int)M, KV_SZ * DH_SZ, D_SZ, 0, 1);
  // Attention: B*H*(T/16) = 4096 waves -> 512 blocks
  attn_kernel<<<512, 256, 0, stream>>>(Qb, Kb, Vt, Ao);
  // Output projection
  out_proj_kernel<<<2048, 256, 0, stream>>>(Ao, Wo, bo, out, (int)M, D_SZ, D_SZ);
}
